// CorrelationMatrix_31387620999841
// MI455X (gfx1250) — matrix operand order repaired and checked by reading, not compiled, not run on hardware
//
#include <hip/hip_runtime.h>
#include <hip/hip_bf16.h>

// ---------------------------------------------------------------------------
// CorrelationMatrix on MI455X (gfx1250, wave32, WMMA, async global->LDS)
//
// Sizes: c=128, q=4096 (64x64), rk=16384 (4 refs * 64x64).
// Pipeline:
//   prep_ft / prep_fr : mask (stride-4 nearest) + f32->bf16, laid out so every
//                       WMMA operand load is two contiguous b128s.
//   gemm1             : corr(q,rk) = ftT x frT   via v_wmma_f32_16x16x32_bf16
//   conv_softmax      : fused conv1(rk-grid 3x3) + conv2(q-grid 3x3) + joint
//                       softmax over rk + cast to bf16 P(q,rk). One WG per q;
//                       double-buffered 2x64KB LDS row staging fed by
//                       global_load_async_to_lds_b128 (ASYNCcnt) so the next
//                       neighbor row streams in while the current one is
//                       convolved.
//   gemm2             : out(c,q) = P x frC       via v_wmma_f32_16x16x32_bf16,
//                       2 c-tiles per wave for A-operand reuse.
// ---------------------------------------------------------------------------

typedef __attribute__((ext_vector_type(16))) __bf16 v16bf;
typedef __attribute__((ext_vector_type(8)))  float  v8f;

union BF16x16 { uint4 u[2]; v16bf v; };

#define C_DIM   128
#define Q_DIM   4096
#define RK_DIM  16384
#define HW      64

__device__ __forceinline__ unsigned short f2bf(float f) {
    unsigned int u = __float_as_uint(f);
    u += 0x7FFFu + ((u >> 16) & 1u);           // round-to-nearest-even
    return (unsigned short)(u >> 16);
}

// ---------------- prep: mask + cast + layout ----------------

__global__ void prep_ft_kernel(const float* __restrict__ feats_t,
                               const float* __restrict__ v_t,
                               unsigned short* __restrict__ ftT /* (q,c) */) {
    int idx = blockIdx.x * blockDim.x + threadIdx.x;       // q*128 + c
    if (idx >= Q_DIM * C_DIM) return;
    int q = idx >> 7, c = idx & 127;
    int ht = q >> 6, wt = q & 63;
    float m = v_t[(ht * 4) * 256 + wt * 4];                // nearest: src = 4*dst
    ftT[idx] = f2bf(feats_t[c * Q_DIM + q] * m);
}

__global__ void prep_fr_kernel(const float* __restrict__ feats_ref,
                               const float* __restrict__ v_ref,
                               unsigned short* __restrict__ frT /* (rk,c) */,
                               unsigned short* __restrict__ frC /* (c,rk) */) {
    int idx = blockIdx.x * blockDim.x + threadIdx.x;       // c*16384 + rk
    if (idx >= C_DIM * RK_DIM) return;
    int rk = idx & (RK_DIM - 1), c = idx >> 14;
    int r = rk >> 12, k = rk & 4095;
    int hr = k >> 6, wr = k & 63;
    float m = v_ref[r * 65536 + (hr * 4) * 256 + wr * 4];
    unsigned short b = f2bf(feats_ref[idx] * m);           // feats_ref flat == c*16384+rk
    frC[idx] = b;
    frT[(size_t)rk * C_DIM + c] = b;
}

// ---------------- GEMM1: corr(q,rk) = ftT(q,c) x frT(rk,c)^T ----------------
// One wave per 16x16 tile, K = 128 (4 WMMA steps).

__global__ void gemm1_kernel(const unsigned short* __restrict__ A,  // ftT (4096 x 128)
                             const unsigned short* __restrict__ B,  // frT (16384 x 128)
                             float* __restrict__ Cm)                // corr (4096 x 16384)
{
    int wave = blockIdx.x * (blockDim.x >> 5) + (threadIdx.x >> 5);
    const int nTilesN = RK_DIM / 16;                    // 1024
    int tm = wave / nTilesN;                            // 0..255
    int tn = wave % nTilesN;
    int lane = threadIdx.x & 31;
    int n = lane & 15, g = lane >> 4;

    const unsigned short* aRow = A + (size_t)(tm * 16 + n) * C_DIM;  // A lane m == lane&15
    const unsigned short* bRow = B + (size_t)(tn * 16 + n) * C_DIM;  // B column n == lane&15

    v8f acc = {};
#pragma unroll
    for (int kb = 0; kb < C_DIM; kb += 32) {
        BF16x16 a, b;
        a.u[0] = *(const uint4*)(aRow + kb + 16 * g);       // K = 16g..16g+7
        a.u[1] = *(const uint4*)(aRow + kb + 16 * g + 8);   // K = 16g+8..16g+15
        b.u[0] = *(const uint4*)(bRow + kb + 16 * g);       // K = 16g..16g+15
        b.u[1] = *(const uint4*)(bRow + kb + 16 * g + 8);
        acc = __builtin_amdgcn_wmma_f32_16x16x32_bf16(
                  false, a.v, false, b.v, (short)0, acc, false, false);
    }
    // D layout: VGPR i -> (M = i + 8g, N = lane&15)
    float* o = Cm + (size_t)(tm * 16) * RK_DIM + tn * 16 + n;
#pragma unroll
    for (int i = 0; i < 8; ++i)
        o[(size_t)(i + 8 * g) * RK_DIM] = acc[i];
}

// ---------------- fused conv1 + conv2 + softmax + bf16 cast ----------------
// One workgroup per q. Registers hold this q's 16384-wide conv2 accumulator
// (16 floats/thread); double-buffered LDS stages neighbor rows via the CDNA5
// async global->LDS path (ASYNCcnt), overlapping DMA with compute.

// Issue one 64KB row load (4 x b128 per thread). Async loads complete in
// order, so "s_wait_asynccnt 4" later means: previous row fully landed.
__device__ __forceinline__ void async_row_load(const float* __restrict__ src,
                                               unsigned ldsByteBase, int tid) {
#pragma unroll
    for (int j = 0; j < 4; ++j) {
        int e = (j << 12) + (tid << 2);                  // float index, 16B/thread
        unsigned loff = ldsByteBase + ((unsigned)e << 2);
        const float* g = src + e;
        asm volatile("global_load_async_to_lds_b128 %0, %1, off"
                     :: "v"(loff), "v"(g) : "memory");
    }
}

__global__ __launch_bounds__(1024)
void conv_softmax_kernel(const float* __restrict__ corr,   // (q, rk)
                         const float* __restrict__ w1, const float* __restrict__ b1,
                         const float* __restrict__ w2, const float* __restrict__ b2,
                         unsigned short* __restrict__ P)   // (q, rk) bf16
{
    extern __shared__ float smem[];                  // 2*16384 rows + 32 red
    float* red = smem + 2 * RK_DIM;

    const int q  = blockIdx.x;
    const int ht = q >> 6, wt = q & 63;
    const int tid = threadIdx.x;

    float w1r[9], w2r[9];
#pragma unroll
    for (int i = 0; i < 9; ++i) { w1r[i] = w1[i]; w2r[i] = w2[i]; }
    const float b1v = b1[0], b2v = b2[0];

    // valid conv2 neighbors (uniform per block: depends only on q)
    int   nbrQ[9];
    float nbrW[9];
    int   nn = 0;
    float w2sum = 0.f;
    for (int dq = 0; dq < 9; ++dq) {
        int h2 = ht + (dq / 3) - 1;
        int w2i = wt + (dq % 3) - 1;
        if ((unsigned)h2 >= (unsigned)HW || (unsigned)w2i >= (unsigned)HW) continue;
        nbrQ[nn] = h2 * HW + w2i;
        nbrW[nn] = w2r[dq];
        w2sum += w2r[dq];
        ++nn;
    }

    float acc[16];
#pragma unroll
    for (int j = 0; j < 16; ++j) acc[j] = 0.f;

    // prologue: start DMA of neighbor 0 into buffer 0 (LDS byte offset 0)
    async_row_load(corr + (size_t)nbrQ[0] * RK_DIM, 0u, tid);

    for (int i = 0; i < nn; ++i) {
        const int cur = i & 1;
        if (i + 1 < nn) {
            // stream next row into the other buffer while we compute
            async_row_load(corr + (size_t)nbrQ[i + 1] * RK_DIM,
                           (unsigned)((cur ^ 1) * RK_DIM * 4), tid);
            asm volatile("s_wait_asynccnt 0x4" ::: "memory");   // row i landed
        } else {
            asm volatile("s_wait_asynccnt 0x0" ::: "memory");
        }
        __syncthreads();                                 // whole buffer visible

        const float* rowp = smem + cur * RK_DIM;
        const float wv = nbrW[i];
        // conv1 over rk-grid (within the staged row), accumulate conv2
#pragma unroll
        for (int j = 0; j < 16; ++j) {
            int rk = tid + (j << 10);
            int rb = rk & ~4095;                         // r*4096
            int kk = rk & 4095;
            int hr = kk >> 6, wr = kk & 63;
            float y1 = 0.f;
#pragma unroll
            for (int di = 0; di < 3; ++di) {
                int h = hr + di - 1;
                if ((unsigned)h >= (unsigned)HW) continue;
#pragma unroll
                for (int dj = 0; dj < 3; ++dj) {
                    int w = wr + dj - 1;
                    if ((unsigned)w >= (unsigned)HW) continue;
                    y1 = fmaf(w1r[di * 3 + dj], rowp[rb + h * HW + w], y1);
                }
            }
            acc[j] = fmaf(wv, y1, acc[j]);
        }
        __syncthreads();     // everyone done reading rowp before its buffer is re-DMA'd
    }

    // fold biases: y2 = sum_valid w2*(raw1 + b1) + b2
    float v[16];
    float lmax = -3.0e38f;
#pragma unroll
    for (int j = 0; j < 16; ++j) {
        v[j] = acc[j] + b1v * w2sum + b2v;
        lmax = fmaxf(lmax, v[j]);
    }

    // joint softmax over the full rk row (wave32 shuffle + LDS across 32 waves)
    const int wid = tid >> 5, lane = tid & 31;
#pragma unroll
    for (int o = 16; o >= 1; o >>= 1) lmax = fmaxf(lmax, __shfl_xor(lmax, o, 32));
    if (lane == 0) red[wid] = lmax;
    __syncthreads();
    if (wid == 0) {
        float t = red[lane];
#pragma unroll
        for (int o = 16; o >= 1; o >>= 1) t = fmaxf(t, __shfl_xor(t, o, 32));
        if (lane == 0) red[0] = t;
    }
    __syncthreads();
    const float mx = red[0];
    __syncthreads();

    float lsum = 0.f;
#pragma unroll
    for (int j = 0; j < 16; ++j) { v[j] = __expf(v[j] - mx); lsum += v[j]; }
#pragma unroll
    for (int o = 16; o >= 1; o >>= 1) lsum += __shfl_xor(lsum, o, 32);
    if (lane == 0) red[wid] = lsum;
    __syncthreads();
    if (wid == 0) {
        float t = red[lane];
#pragma unroll
        for (int o = 16; o >= 1; o >>= 1) t += __shfl_xor(t, o, 32);
        if (lane == 0) red[0] = t;
    }
    __syncthreads();
    const float inv = 1.0f / red[0];

    unsigned short* prow = P + (size_t)q * RK_DIM;
#pragma unroll
    for (int j = 0; j < 16; ++j)
        prow[tid + (j << 10)] = f2bf(v[j] * inv);
}

// ---------------- GEMM2: out(c,q) = P(q,rk) x frC(c,rk)^T ----------------
// One wave per 16(M) x 32(N) tile pair: K = 16384 (512 WMMA steps each),
// A operand loaded once and reused for both c-tiles.

__global__ void gemm2_kernel(const unsigned short* __restrict__ A,  // P   (4096 x 16384)
                             const unsigned short* __restrict__ B,  // frC (128 x 16384)
                             float* __restrict__ Out)               // (128 x 4096)
{
    int wave = blockIdx.x * (blockDim.x >> 5) + (threadIdx.x >> 5);
    const int nPairsN = C_DIM / 32;                     // 4 pairs of c-tiles
    int tm = wave / nPairsN;                            // 0..255 (q tiles)
    int tp = wave % nPairsN;                            // 0..3   (c tile pairs)
    int lane = threadIdx.x & 31;
    int n = lane & 15, g = lane >> 4;

    const unsigned short* aRow  = A + (size_t)(tm * 16 + n) * RK_DIM;
    const unsigned short* bRow0 = B + (size_t)(tp * 32 + n) * RK_DIM;
    const unsigned short* bRow1 = bRow0 + (size_t)16 * RK_DIM;

    v8f acc0 = {}, acc1 = {};
    for (int kb = 0; kb < RK_DIM; kb += 32) {
        __builtin_prefetch(aRow + kb + 512, 0, 0);       // global_prefetch_b8
        BF16x16 a, b0, b1;
        a.u[0]  = *(const uint4*)(aRow + kb + 16 * g);
        a.u[1]  = *(const uint4*)(aRow + kb + 16 * g + 8);
        b0.u[0] = *(const uint4*)(bRow0 + kb + 16 * g);
        b0.u[1] = *(const uint4*)(bRow0 + kb + 16 * g + 8);
        b1.u[0] = *(const uint4*)(bRow1 + kb + 16 * g);
        b1.u[1] = *(const uint4*)(bRow1 + kb + 16 * g + 8);
        acc0 = __builtin_amdgcn_wmma_f32_16x16x32_bf16(
                   false, a.v, false, b0.v, (short)0, acc0, false, false);
        acc1 = __builtin_amdgcn_wmma_f32_16x16x32_bf16(
                   false, a.v, false, b1.v, (short)0, acc1, false, false);
    }
    // out[(c)*4096 + q]; per lane the 8 M-values are contiguous in q
    float* o0 = Out + (size_t)(tp * 32 + n) * Q_DIM + tm * 16;
    float* o1 = o0 + (size_t)16 * Q_DIM;
#pragma unroll
    for (int i = 0; i < 8; ++i) {
        o0[i + 8 * g] = acc0[i];
        o1[i + 8 * g] = acc1[i];
    }
}

// ---------------------------------------------------------------------------

extern "C" void kernel_launch(void* const* d_in, const int* in_sizes, int n_in,
                              void* d_out, int out_size, void* d_ws, size_t ws_size,
                              hipStream_t stream) {
    const float* feats_t   = (const float*)d_in[0];   // (1,128,64,64)
    const float* feats_ref = (const float*)d_in[1];   // (1,128,4,64,64)
    const float* v_t       = (const float*)d_in[2];   // (1,1,256,256)
    const float* v_ref     = (const float*)d_in[3];   // (1,1,4,256,256)
    const float* conv1_w   = (const float*)d_in[4];
    const float* conv1_b   = (const float*)d_in[5];
    const float* conv2_w   = (const float*)d_in[6];
    const float* conv2_b   = (const float*)d_in[7];
    float* out = (float*)d_out;

    // workspace carve-up (~412 MB)
    char* ws = (char*)d_ws;
    float*          corr0 = (float*)ws;                                  // 268435456 B
    unsigned short* P     = (unsigned short*)(ws + 268435456ULL);        // 134217728 B
    unsigned short* ftT   = (unsigned short*)(ws + 402653184ULL);        //   1048576 B
    unsigned short* frT   = (unsigned short*)(ws + 403701760ULL);        //   4194304 B
    unsigned short* frC   = (unsigned short*)(ws + 407896064ULL);        //   4194304 B

    prep_ft_kernel<<<(Q_DIM * C_DIM) / 256, 256, 0, stream>>>(feats_t, v_t, ftT);
    prep_fr_kernel<<<(C_DIM * RK_DIM) / 256, 256, 0, stream>>>(feats_ref, v_ref, frT, frC);

    // 256 x 1024 tiles, 4 waves per 128-thread block
    gemm1_kernel<<<(256 * 1024) / 4, 128, 0, stream>>>(ftT, frT, corr0);

    // 2*64KB row double-buffer + 32-float reduction scratch (dynamic LDS)
    const size_t convShmem = (2 * RK_DIM + 32) * sizeof(float);
    conv_softmax_kernel<<<Q_DIM, 1024, convShmem, stream>>>(corr0, conv1_w, conv1_b,
                                                            conv2_w, conv2_b, P);

    // 256 x 4 tile-pairs, 4 waves per 128-thread block
    gemm2_kernel<<<(256 * 4) / 4, 128, 0, stream>>>(P, frC, out);
}
